// MultiHeadAttention_29961691857645
// MI455X (gfx1250) — matrix operand order repaired and checked by reading, not compiled, not run on hardware
//
#include <hip/hip_runtime.h>
#include <math.h>
#include <stdint.h>

#define BB  2
#define SS  2048
#define DD  1024
#define HH  16
#define DKK 64

typedef __attribute__((ext_vector_type(16))) __bf16 v16bf;
typedef __attribute__((ext_vector_type(8)))  __bf16 v8bf;
typedef __attribute__((ext_vector_type(8)))  float  v8f;

union BF16x16 { v16bf v; v8bf h[2]; };

__device__ __forceinline__ v16bf load_frag2(const __bf16* p0, const __bf16* p1) {
  BF16x16 u;
  u.h[0] = *(const v8bf*)p0;
  u.h[1] = *(const v8bf*)p1;
  return u.v;
}

// A-fragment: 16x32 bf16 tile at (mbase, kbase), row stride lda.
// lanes 0-15: row M=r, k {0..7,16..23}; lanes 16-31: k {8..15,24..31}.
__device__ __forceinline__ v16bf load_a_frag(const __bf16* A, int lda, int mbase,
                                             int kbase, int r, int half) {
  const __bf16* rp = A + (size_t)(mbase + r) * lda + kbase + half * 8;
  return load_frag2(rp, rp + 16);
}

// B-fragment for Bmat[k][n] = W[n][k], W row-major (N,K) stride ldw.
// lane = column n; lanes 0-15 hold K {0..7,16..23}, lanes 16-31 K {8..15,24..31}.
__device__ __forceinline__ v16bf load_bT_frag(const __bf16* W, int ldw, int nbase,
                                              int kbase, int r, int half) {
  const __bf16* rp = W + (size_t)(nbase + r) * ldw + kbase + half * 8;
  return load_frag2(rp, rp + 16);
}

__device__ __forceinline__ v8f wmma_bf16(v16bf a, v16bf b, v8f c) {
  return __builtin_amdgcn_wmma_f32_16x16x32_bf16(false, a, false, b,
                                                 (short)0, c, false, false);
}

// CDNA5 async global->LDS copy (ASYNCcnt-tracked DMA, 512B per wave-instr)
__device__ __forceinline__ void async_copy_b128(uint32_t lds_addr, uint64_t gaddr) {
  asm volatile("global_load_async_to_lds_b128 %0, %1, off"
               :: "v"(lds_addr), "v"(gaddr) : "memory");
}

// ---------------------------------------------------------------- fp32->bf16
__global__ void cvt_f32_bf16(const float* __restrict__ src, __bf16* __restrict__ dst, int n) {
  int i = (blockIdx.x * blockDim.x + threadIdx.x) * 4;
  if (i + 3 < n) {
    float4 f = *(const float4*)(src + i);
    dst[i + 0] = (__bf16)f.x;
    dst[i + 1] = (__bf16)f.y;
    dst[i + 2] = (__bf16)f.z;
    dst[i + 3] = (__bf16)f.w;
  }
}

// ------------------------------------------------- projection GEMM: y = x W^T
// Software-pipelined: fragments for k+32 are loaded while k's WMMAs execute.
__global__ __launch_bounds__(128) void gemm_qkv(const __bf16* __restrict__ X,
                                                const __bf16* __restrict__ W,
                                                __bf16* __restrict__ Out, int vmode) {
  const int lane = threadIdx.x & 31;
  const int r = lane & 15, half = lane >> 4;
  const int wave = blockIdx.x * (blockDim.x >> 5) + (threadIdx.x >> 5);
  const int mbase = (wave >> 4) * 16;   // 256 m-tiles
  const int nbase = (wave & 15) * 64;   // 16 n-strips

  v16bf a  = load_a_frag(X, DD, mbase, 0, r, half);
  v16bf b0 = load_bT_frag(W, DD, nbase +  0, 0, r, half);
  v16bf b1 = load_bT_frag(W, DD, nbase + 16, 0, r, half);
  v16bf b2 = load_bT_frag(W, DD, nbase + 32, 0, r, half);
  v16bf b3 = load_bT_frag(W, DD, nbase + 48, 0, r, half);

  v8f acc[4] = {};
  for (int kb = 0; kb < DD; kb += 32) {
    const int kn = (kb + 32 < DD) ? kb + 32 : 0;       // harmless reload on last iter
    v16bf an  = load_a_frag(X, DD, mbase, kn, r, half);
    v16bf bn0 = load_bT_frag(W, DD, nbase +  0, kn, r, half);
    v16bf bn1 = load_bT_frag(W, DD, nbase + 16, kn, r, half);
    v16bf bn2 = load_bT_frag(W, DD, nbase + 32, kn, r, half);
    v16bf bn3 = load_bT_frag(W, DD, nbase + 48, kn, r, half);
    acc[0] = wmma_bf16(a, b0, acc[0]);
    acc[1] = wmma_bf16(a, b1, acc[1]);
    acc[2] = wmma_bf16(a, b2, acc[2]);
    acc[3] = wmma_bf16(a, b3, acc[3]);
    a = an; b0 = bn0; b1 = bn1; b2 = bn2; b3 = bn3;
  }
#pragma unroll
  for (int t = 0; t < 4; ++t) {
    const int n = nbase + t * 16 + r;
    const int h = n >> 6, dk = n & 63;
#pragma unroll
    for (int i = 0; i < 8; ++i) {
      const int m = mbase + i + 8 * half;              // C layout: VGPR i -> row i+8*half
      const int bb = m >> 11, s = m & (SS - 1);
      size_t idx;
      if (vmode == 0) idx = (((size_t)(bb * HH + h) * SS) + s) * DKK + dk;   // (B,H,S,DK)
      else            idx = (((size_t)(bb * HH + h) * DKK) + dk) * SS + s;   // (B,H,DK,S)
      Out[idx] = (__bf16)acc[t][i];
    }
  }
}

// ------------------------------------------------- output GEMM: out = A Wo^T (fp32)
__global__ __launch_bounds__(128) void gemm_out(const __bf16* __restrict__ A,
                                                const __bf16* __restrict__ W,
                                                float* __restrict__ Out) {
  const int lane = threadIdx.x & 31;
  const int r = lane & 15, half = lane >> 4;
  const int wave = blockIdx.x * (blockDim.x >> 5) + (threadIdx.x >> 5);
  const int mbase = (wave >> 4) * 16;
  const int nbase = (wave & 15) * 64;

  v16bf a  = load_a_frag(A, DD, mbase, 0, r, half);
  v16bf b0 = load_bT_frag(W, DD, nbase +  0, 0, r, half);
  v16bf b1 = load_bT_frag(W, DD, nbase + 16, 0, r, half);
  v16bf b2 = load_bT_frag(W, DD, nbase + 32, 0, r, half);
  v16bf b3 = load_bT_frag(W, DD, nbase + 48, 0, r, half);

  v8f acc[4] = {};
  for (int kb = 0; kb < DD; kb += 32) {
    const int kn = (kb + 32 < DD) ? kb + 32 : 0;
    v16bf an  = load_a_frag(A, DD, mbase, kn, r, half);
    v16bf bn0 = load_bT_frag(W, DD, nbase +  0, kn, r, half);
    v16bf bn1 = load_bT_frag(W, DD, nbase + 16, kn, r, half);
    v16bf bn2 = load_bT_frag(W, DD, nbase + 32, kn, r, half);
    v16bf bn3 = load_bT_frag(W, DD, nbase + 48, kn, r, half);
    acc[0] = wmma_bf16(a, b0, acc[0]);
    acc[1] = wmma_bf16(a, b1, acc[1]);
    acc[2] = wmma_bf16(a, b2, acc[2]);
    acc[3] = wmma_bf16(a, b3, acc[3]);
    a = an; b0 = bn0; b1 = bn1; b2 = bn2; b3 = bn3;
  }
#pragma unroll
  for (int t = 0; t < 4; ++t) {
    const int n = nbase + t * 16 + r;
#pragma unroll
    for (int i = 0; i < 8; ++i) {
      const int m = mbase + i + 8 * half;
      Out[(size_t)m * DD + n] = acc[t][i];
    }
  }
}

// ---- async staging of K tile: 32 keys x 64 dk bf16 = one contiguous 4KB block
__device__ __forceinline__ void stage_k(const __bf16* Kp, int kb, __bf16* dst, int lane) {
  uint64_t g = (uint64_t)(uintptr_t)(Kp + (size_t)kb * DKK) + (uint32_t)(lane * 16);
  uint32_t l = (uint32_t)(uintptr_t)dst + (uint32_t)(lane * 16);
#pragma unroll
  for (int j = 0; j < 8; ++j) async_copy_b128(l + j * 512, g + j * 512);
}

// ---- async staging of V tile: Vt is (DK,S); 64 rows x 64B, row stride SS*2 bytes
__device__ __forceinline__ void stage_v(const __bf16* Vp, int kb, __bf16* dst, int lane) {
  const int seg = lane & 3, rw = lane >> 2;            // 8 rows per instruction
  uint64_t g = (uint64_t)(uintptr_t)(Vp + (size_t)rw * SS + kb) + (uint32_t)(seg * 16);
  uint32_t l = (uint32_t)(uintptr_t)dst + (uint32_t)(rw * 64 + seg * 16);
#pragma unroll
  for (int j = 0; j < 8; ++j)
    async_copy_b128(l + j * (8 * 64), g + (uint64_t)j * (8 * SS * 2));
}

// ------------------------------------------------- flash attention
// One wave per (b,h,16-row q tile). K/V tiles double-buffered in LDS via
// global_load_async_to_lds_b128 (ASYNCcnt), online softmax, P transposed via LDS.
__global__ __launch_bounds__(128) void attn_kernel(const __bf16* __restrict__ Qh,
                                                   const __bf16* __restrict__ Kh,
                                                   const __bf16* __restrict__ Vt,
                                                   const int* __restrict__ mask,
                                                   __bf16* __restrict__ AttnOut) {
  // per-wave: K 2x4KB + V 2x4KB + P 1KB = 17KB; 4 waves -> 68KB of the WGP's 320KB
  __shared__ __align__(16) char lds[4 * (4 * 4096 + 1024)];
  const int lane = threadIdx.x & 31;
  const int r = lane & 15, half = lane >> 4;
  char* wbase = lds + ((threadIdx.x >> 5) & 3) * (4 * 4096 + 1024);
  __bf16* Kb0 = (__bf16*)(wbase);
  __bf16* Kb1 = (__bf16*)(wbase + 4096);
  __bf16* Vb0 = (__bf16*)(wbase + 8192);
  __bf16* Vb1 = (__bf16*)(wbase + 12288);
  __bf16* P   = (__bf16*)(wbase + 16384);

  const int wave  = blockIdx.x * (blockDim.x >> 5) + (threadIdx.x >> 5);
  const int qbase = (wave & (SS / 16 - 1)) * 16;   // 128 q-tiles
  const int bh    = wave >> 7;                     // 0..31
  const int bb    = bh >> 4, h = bh & 15;

  const __bf16* Qp = Qh + (size_t)bh * SS * DKK;
  const __bf16* Kp = Kh + (size_t)bh * SS * DKK;
  const __bf16* Vp = Vt + (size_t)bh * DKK * SS;
  const int*    Mp = mask + (size_t)bb * SS * SS;

  // Q tile (16x64) pinned in registers: two A-fragments.
  v16bf qa0 = load_a_frag(Qp, DKK, qbase, 0,  r, half);
  v16bf qa1 = load_a_frag(Qp, DKK, qbase, 32, r, half);

  v8f o0 = {}, o1 = {}, o2 = {}, o3 = {};
  float mi[8], li[8];
#pragma unroll
  for (int i = 0; i < 8; ++i) { mi[i] = -INFINITY; li[i] = 0.f; }

  // prologue: stage tile 0 (16 async instrs outstanding)
  stage_k(Kp, 0, Kb0, lane);
  stage_v(Vp, 0, Vb0, lane);

  int cur = 0;
  for (int kb = 0; kb < SS; kb += 32) {
    const bool hasNext = (kb + 32) < SS;
    // WAR guard: prior ds reads of the buffer we are about to overwrite are done
    asm volatile("s_wait_dscnt 0x0" ::: "memory");
    if (hasNext) {
      stage_k(Kp, kb + 32, cur ? Kb0 : Kb1, lane);
      stage_v(Vp, kb + 32, cur ? Vb0 : Vb1, lane);
      asm volatile("s_wait_asynccnt 0x10" ::: "memory");  // tile `kb` landed; 16 in flight
    } else {
      asm volatile("s_wait_asynccnt 0x0" ::: "memory");
    }
    const __bf16* Kt = cur ? Kb1 : Kb0;
    const __bf16* Vv = cur ? Vb1 : Vb0;

    // scores S[16x32] = Q (16x64) * K^T, B-fragments from LDS (ds_load_b128)
    v8f s0 = {}, s1 = {};
    s0 = wmma_bf16(qa0, load_bT_frag(Kt, DKK, 0,  0,  r, half), s0);
    s0 = wmma_bf16(qa1, load_bT_frag(Kt, DKK, 0,  32, r, half), s0);
    s1 = wmma_bf16(qa0, load_bT_frag(Kt, DKK, 16, 0,  r, half), s1);
    s1 = wmma_bf16(qa1, load_bT_frag(Kt, DKK, 16, 32, r, half), s1);

    const int n0 = kb + r, n1 = kb + 16 + r;
    float p0[8], p1[8];
#pragma unroll
    for (int i = 0; i < 8; ++i) {
      const int m = qbase + i + 8 * half;
      float v0 = s0[i] * 0.125f;                       // 1/sqrt(64)
      float v1 = s1[i] * 0.125f;
      if (Mp[(size_t)m * SS + n0] == 0) v0 = -1e9f;
      if (Mp[(size_t)m * SS + n1] == 0) v1 = -1e9f;
      p0[i] = v0; p1[i] = v1;
    }
    // online softmax; row stats live replicated across each 16-lane half
#pragma unroll
    for (int i = 0; i < 8; ++i) {
      float mx = fmaxf(p0[i], p1[i]);
      mx = fmaxf(mx, __shfl_xor(mx, 1, 32));
      mx = fmaxf(mx, __shfl_xor(mx, 2, 32));
      mx = fmaxf(mx, __shfl_xor(mx, 4, 32));
      mx = fmaxf(mx, __shfl_xor(mx, 8, 32));
      const float mnew = fmaxf(mi[i], mx);
      const float corr = __expf(mi[i] - mnew);
      mi[i] = mnew;
      const float e0 = __expf(p0[i] - mnew);
      const float e1 = __expf(p1[i] - mnew);
      float rs = e0 + e1;
      rs += __shfl_xor(rs, 1, 32);
      rs += __shfl_xor(rs, 2, 32);
      rs += __shfl_xor(rs, 4, 32);
      rs += __shfl_xor(rs, 8, 32);
      li[i] = li[i] * corr + rs;
      o0[i] *= corr; o1[i] *= corr; o2[i] *= corr; o3[i] *= corr;
      p0[i] = e0; p1[i] = e1;
    }
    // P: C-layout -> row-major 16x32 bf16 in LDS -> reload in A-fragment layout
#pragma unroll
    for (int i = 0; i < 8; ++i) {
      P[(i + 8 * half) * 32 + r]      = (__bf16)p0[i];
      P[(i + 8 * half) * 32 + 16 + r] = (__bf16)p1[i];
    }
    asm volatile("s_wait_dscnt 0x0" ::: "memory");
    __builtin_amdgcn_wave_barrier();
    v16bf pa = load_a_frag(P, 32, 0, 0, r, half);
    __builtin_amdgcn_wave_barrier();

    // O[16x64] += P (16x32) * V (32x64); V tile row-major (dk, key) in LDS
    o0 = wmma_bf16(pa, load_bT_frag(Vv, 32, 0,  0, r, half), o0);
    o1 = wmma_bf16(pa, load_bT_frag(Vv, 32, 16, 0, r, half), o1);
    o2 = wmma_bf16(pa, load_bT_frag(Vv, 32, 32, 0, r, half), o2);
    o3 = wmma_bf16(pa, load_bT_frag(Vv, 32, 48, 0, r, half), o3);
    cur ^= 1;
  }

  // epilogue: O /= l, write (B,S,D) bf16 for the output projection
#pragma unroll
  for (int i = 0; i < 8; ++i) {
    const int m = qbase + i + 8 * half;
    const float inv = 1.0f / li[i];
    size_t base = ((size_t)(bb * SS + m)) * DD + h * DKK + r;
    AttnOut[base]      = (__bf16)(o0[i] * inv);
    AttnOut[base + 16] = (__bf16)(o1[i] * inv);
    AttnOut[base + 32] = (__bf16)(o2[i] * inv);
    AttnOut[base + 48] = (__bf16)(o3[i] * inv);
  }
}

// ---------------------------------------------------------------- launcher
extern "C" void kernel_launch(void* const* d_in, const int* in_sizes, int n_in,
                              void* d_out, int out_size, void* d_ws, size_t ws_size,
                              hipStream_t stream) {
  const float* q    = (const float*)d_in[0];
  const float* k    = (const float*)d_in[1];
  const float* v    = (const float*)d_in[2];
  const int*   mask = (const int*)d_in[3];
  const float* Wq   = (const float*)d_in[4];
  const float* Wk   = (const float*)d_in[5];
  const float* Wv   = (const float*)d_in[6];
  const float* Wo   = (const float*)d_in[7];

  const size_t NX = (size_t)BB * SS * DD;   // 4M elements
  const size_t NW = (size_t)DD * DD;        // 1M elements

  char* ws = (char*)d_ws;
  size_t off = 0;
  auto carve = [&](size_t elems) { __bf16* p = (__bf16*)(ws + off); off += elems * sizeof(__bf16); return p; };
  __bf16* Xq  = carve(NX);
  __bf16* Xk  = carve(NX);
  __bf16* Xv  = carve(NX);
  __bf16* Wqb = carve(NW);
  __bf16* Wkb = carve(NW);
  __bf16* Wvb = carve(NW);
  __bf16* Wob = carve(NW);
  __bf16* Qhb = carve(NX);     // (B,H,S,DK)
  __bf16* Khb = carve(NX);     // (B,H,S,DK)
  __bf16* Vtb = carve(NX);     // (B,H,DK,S)
  __bf16* Ab  = Xq;            // attn output (B,S,D); Xq dead by then

  // fp32 -> bf16
  cvt_f32_bf16<<<(int)(NX / 1024), 256, 0, stream>>>(q,  Xq,  (int)NX);
  cvt_f32_bf16<<<(int)(NX / 1024), 256, 0, stream>>>(k,  Xk,  (int)NX);
  cvt_f32_bf16<<<(int)(NX / 1024), 256, 0, stream>>>(v,  Xv,  (int)NX);
  cvt_f32_bf16<<<(int)(NW / 1024), 256, 0, stream>>>(Wq, Wqb, (int)NW);
  cvt_f32_bf16<<<(int)(NW / 1024), 256, 0, stream>>>(Wk, Wkb, (int)NW);
  cvt_f32_bf16<<<(int)(NW / 1024), 256, 0, stream>>>(Wv, Wvb, (int)NW);
  cvt_f32_bf16<<<(int)(NW / 1024), 256, 0, stream>>>(Wo, Wob, (int)NW);

  // projections: 4096 waves = 1024 blocks x 4 waves
  gemm_qkv<<<1024, 128, 0, stream>>>(Xq, Wqb, Qhb, 0);
  gemm_qkv<<<1024, 128, 0, stream>>>(Xk, Wkb, Khb, 0);
  gemm_qkv<<<1024, 128, 0, stream>>>(Xv, Wvb, Vtb, 1);

  // attention: one wave per (b,h,qtile) = 4096 waves
  attn_kernel<<<1024, 128, 0, stream>>>(Qhb, Khb, Vtb, mask, Ab);

  // output projection -> fp32
  gemm_out<<<1024, 128, 0, stream>>>(Ab, Wob, (float*)d_out);
}
